// LocalAttention_55473797595221
// MI455X (gfx1250) — compile-verified, matrix operand order checked by reading, not run on hardware
//
#include <hip/hip_runtime.h>
#include <hip/hip_bf16.h>

#define L_SEQ 6016
#define PAD   128
#define LPAD  6144
#define DM    1024
#define DL    512
#define NW    24
#define GW    768   // 3 * L_WIN keys per group

typedef __attribute__((ext_vector_type(16))) __bf16 bfrag_t;
typedef __attribute__((ext_vector_type(8)))  float  v8f;

union FragB { bfrag_t v; uint4 q[2]; };
union AccF  { v8f v; float f[8]; };

__device__ __forceinline__ unsigned short f32_bf16(float x) {
    unsigned int u = __float_as_uint(x);
    u += 0x7FFFu + ((u >> 16) & 1u);          // round-to-nearest-even
    return (unsigned short)(u >> 16);
}

// Load a 16x32 bf16 WMMA fragment. Lane l: row/col = l%16, hi = l/16.
// K elements: {hi*8 + 0..7, 16 + hi*8 + 0..7}, each run 16B -> two b128 loads.
__device__ __forceinline__ bfrag_t load_frag(const unsigned short* base, int stride, int lane) {
    const int r  = lane & 15;
    const int hi = lane >> 4;
    const unsigned short* p = base + r * stride + hi * 8;
    FragB f;
    f.q[0] = *(const uint4*)(p);
    f.q[1] = *(const uint4*)(p + 16);
    return f.v;
}

__device__ __forceinline__ v8f wmma_bf16(bfrag_t a, bfrag_t b, v8f c) {
    return __builtin_amdgcn_wmma_f32_16x16x32_bf16(false, a, false, b, (short)0, c, false, false);
}

// C = X(pad)[M x 1024] * W[1024 x 512] + bias  ->  bf16 out  (double-buffered LDS)
// transposed==0: out[(b*6144 + t)*512 + n]; transposed==1: out[(b*512 + n)*6144 + t]
__global__ __launch_bounds__(256)
void proj_kernel(const float* __restrict__ X, const float* __restrict__ W,
                 const float* __restrict__ bias, unsigned short* __restrict__ out,
                 int transposed)
{
    __shared__ unsigned short As[2][64 * 32];   // [row][k]
    __shared__ unsigned short Bs[2][64 * 32];   // [n][k]  (transposed)

    const int tid  = threadIdx.x;
    const int lane = tid & 31;
    const int wv   = tid >> 5;
    const int wm   = wv >> 1;                // 0..3  (16-row subtile)
    const int wn   = wv & 1;                 // 0..1  (32-col subtile)
    const int tileM = blockIdx.x;
    const int tileN = blockIdx.y;

    // A staging: 64 rows x 32 cols, 8 floats/thread
    const int ar = tid >> 2;                 // 0..63
    const int ac = (tid & 3) << 3;           // 0,8,16,24
    const int rowg  = tileM * 64 + ar;
    const int bidx  = rowg / LPAD;
    const int tp    = rowg % LPAD;
    const int realr = tp - PAD;
    const float* Xrow = X + ((long long)bidx * L_SEQ + realr) * DM;

    // B staging: 32 k-rows x 64 cols
    const int bkr = tid >> 3;                // 0..31
    const int bnc = (tid & 7) << 3;          // 0..56

    auto stage = [&](int kk, int bufi) {
        unsigned short* da = As[bufi] + ar * 32 + ac;
        if (realr >= 0) {
            float4 x0 = *(const float4*)(Xrow + kk + ac);
            float4 x1 = *(const float4*)(Xrow + kk + ac + 4);
            da[0] = f32_bf16(x0.x); da[1] = f32_bf16(x0.y);
            da[2] = f32_bf16(x0.z); da[3] = f32_bf16(x0.w);
            da[4] = f32_bf16(x1.x); da[5] = f32_bf16(x1.y);
            da[6] = f32_bf16(x1.z); da[7] = f32_bf16(x1.w);
        } else {
            for (int i = 0; i < 8; ++i) da[i] = 0;
        }
        const float* wr = W + (long long)(kk + bkr) * DL + tileN * 64 + bnc;
        float4 w0 = *(const float4*)(wr);
        float4 w1 = *(const float4*)(wr + 4);
        float wf[8] = {w0.x, w0.y, w0.z, w0.w, w1.x, w1.y, w1.z, w1.w};
        for (int i = 0; i < 8; ++i)
            Bs[bufi][(bnc + i) * 32 + bkr] = f32_bf16(wf[i]);
    };

    AccF c0, c1;
    for (int i = 0; i < 8; ++i) { c0.f[i] = 0.f; c1.f[i] = 0.f; }

    stage(0, 0);
    const int nsteps = DM / 32;
    for (int i = 0; i < nsteps; ++i) {
        const int cur = i & 1;
        __syncthreads();
        if (i + 1 < nsteps) stage((i + 1) * 32, cur ^ 1);
        bfrag_t a  = load_frag(As[cur] + (wm * 16) * 32, 32, lane);
        bfrag_t b0 = load_frag(Bs[cur] + (wn * 32 + 0)  * 32, 32, lane);
        bfrag_t b1 = load_frag(Bs[cur] + (wn * 32 + 16) * 32, 32, lane);
        c0.v = wmma_bf16(a, b0, c0.v);
        c1.v = wmma_bf16(a, b1, c1.v);
    }

    const int hi = lane >> 4, ln = lane & 15;
    for (int sub = 0; sub < 2; ++sub) {
        AccF& cc = sub ? c1 : c0;
        const int n  = tileN * 64 + wn * 32 + sub * 16 + ln;
        const float bv = bias[n];
        for (int r = 0; r < 8; ++r) {
            const int rg  = tileM * 64 + wm * 16 + r + 8 * hi;
            const int bb  = rg / LPAD;
            const int tpp = rg % LPAD;
            const unsigned short val = f32_bf16(cc.f[r] + bv);
            if (transposed)
                out[((long long)bb * DL + n) * LPAD + tpp] = val;
            else
                out[((long long)bb * LPAD + tpp) * DL + n] = val;
        }
    }
}

// One block per (batch, window, 32-query-row block): S = QK^T, softmax, O = P V
__global__ __launch_bounds__(256)
void attn_kernel(const unsigned short* __restrict__ pq,
                 const unsigned short* __restrict__ pk,
                 const unsigned short* __restrict__ pvT,
                 unsigned short* __restrict__ po)
{
    __shared__ float          S[32][GW];     // 96 KB scores
    __shared__ unsigned short P[32][GW];     // 48 KB probabilities (bf16, unnormalized)
    __shared__ float          red[32][8];
    __shared__ float          invs[32];

    const int tid  = threadIdx.x;
    const int lane = tid & 31;
    const int wv   = tid >> 5;
    const int rb = blockIdx.x;               // 0..7
    const int w  = blockIdx.y;               // 0..23
    const int b  = blockIdx.z;
    const int g  = w / 3;

    const unsigned short* qbase = pq  + ((long long)b * LPAD + w * 256 + rb * 32) * DL;
    const unsigned short* kbase = pk  + ((long long)b * LPAD + g * GW) * DL;
    const unsigned short* vbase = pvT + (long long)b * DL * LPAD + g * GW; // [f][t]

    const int hi = lane >> 4, ln = lane & 15;
    const float scale = 0.04419417382f;      // 1/sqrt(512)

    // ---- Phase 1: S = (Q K^T) * scale ; paired key tiles reuse A fragment ----
    for (int p = wv; p < 48; p += 8) {
        const int tm  = p & 1;               // query 16-row half
        const int tn0 = (p >> 1) * 2;        // first of two key 16-tiles
        AccF c0, c1;
        for (int i = 0; i < 8; ++i) { c0.f[i] = 0.f; c1.f[i] = 0.f; }
        const unsigned short* ab  = qbase + (tm * 16) * DL;
        const unsigned short* bb0 = kbase + (tn0 * 16) * DL;
        const unsigned short* bb1 = bb0 + 16 * DL;
        for (int kk = 0; kk < DL; kk += 32) {
            bfrag_t a  = load_frag(ab  + kk, DL, lane);
            bfrag_t f0 = load_frag(bb0 + kk, DL, lane);
            bfrag_t f1 = load_frag(bb1 + kk, DL, lane);
            c0.v = wmma_bf16(a, f0, c0.v);
            c1.v = wmma_bf16(a, f1, c1.v);
        }
        for (int r = 0; r < 8; ++r) {
            const int m = tm * 16 + r + 8 * hi;
            S[m][(tn0 + 0) * 16 + ln] = c0.f[r] * scale;
            S[m][(tn0 + 1) * 16 + ln] = c1.f[r] * scale;
        }
    }
    __syncthreads();

    // ---- Phase 2: row max, single exp pass (unnormalized P), deferred 1/sum ----
    const int row = tid >> 3;
    const int sub = tid & 7;
    float mx = -3.0e38f;
    for (int cix = sub; cix < GW; cix += 8) mx = fmaxf(mx, S[row][cix]);
    red[row][sub] = mx;
    __syncthreads();
    float rm = red[row][0];
    for (int i = 1; i < 8; ++i) rm = fmaxf(rm, red[row][i]);
    __syncthreads();
    float sum = 0.f;
    for (int cix = sub; cix < GW; cix += 8) {
        float e = __expf(S[row][cix] - rm);
        sum += e;
        P[row][cix] = f32_bf16(e);
    }
    red[row][sub] = sum;
    __syncthreads();
    if (sub == 0) {
        float rs = 0.f;
        for (int i = 0; i < 8; ++i) rs += red[row][i];
        invs[row] = 1.0f / rs;
    }
    __syncthreads();

    // ---- Phase 3: O = (P @ V) * inv ; paired feature tiles reuse A fragment ----
    unsigned short* obase = po + ((long long)b * LPAD + w * 256 + rb * 32) * DL;
    for (int p = wv; p < 32; p += 8) {
        const int tm  = p & 1;
        const int tf0 = (p >> 1) * 2;        // first of two feature 16-tiles
        AccF c0, c1;
        for (int i = 0; i < 8; ++i) { c0.f[i] = 0.f; c1.f[i] = 0.f; }
        const unsigned short* bb0 = vbase + (long long)(tf0 * 16) * LPAD;
        const unsigned short* bb1 = bb0 + 16 * LPAD;
        for (int kk = 0; kk < GW; kk += 32) {
            bfrag_t a  = load_frag(&P[tm * 16][0] + kk, GW, lane);
            bfrag_t f0 = load_frag(bb0 + kk, LPAD, lane);
            bfrag_t f1 = load_frag(bb1 + kk, LPAD, lane);
            c0.v = wmma_bf16(a, f0, c0.v);
            c1.v = wmma_bf16(a, f1, c1.v);
        }
        for (int r = 0; r < 8; ++r) {
            const int m = tm * 16 + r + 8 * hi;
            const float iv = invs[m];
            obase[(long long)m * DL + (tf0 + 0) * 16 + ln] = f32_bf16(c0.f[r] * iv);
            obase[(long long)m * DL + (tf0 + 1) * 16 + ln] = f32_bf16(c1.f[r] * iv);
        }
    }
}

// out = o[bf16, M x 512] * Wo[512 x 1024] + bo  (drop left-pad rows), f32 out
__global__ __launch_bounds__(256)
void outproj_kernel(const unsigned short* __restrict__ o,
                    const float* __restrict__ Wo,
                    const float* __restrict__ bo,
                    float* __restrict__ out)
{
    __shared__ unsigned short Bs[2][64 * 32];   // [n][k]
    const int tid = threadIdx.x, lane = tid & 31, wv = tid >> 5;
    const int wm = wv >> 1, wn = wv & 1;
    const int tileM = blockIdx.x, tileN = blockIdx.y;
    const int bkr = tid >> 3, bnc = (tid & 7) << 3;

    auto stageB = [&](int kk, int bufi) {
        const float* wr = Wo + (long long)(kk + bkr) * DM + tileN * 64 + bnc;
        float4 w0 = *(const float4*)(wr);
        float4 w1 = *(const float4*)(wr + 4);
        float wf[8] = {w0.x, w0.y, w0.z, w0.w, w1.x, w1.y, w1.z, w1.w};
        for (int i = 0; i < 8; ++i)
            Bs[bufi][(bnc + i) * 32 + bkr] = f32_bf16(wf[i]);
    };

    AccF c0, c1;
    for (int i = 0; i < 8; ++i) { c0.f[i] = 0.f; c1.f[i] = 0.f; }

    const unsigned short* abase = o + (long long)(tileM * 64 + wm * 16) * DL;

    stageB(0, 0);
    const int nsteps = DL / 32;
    for (int i = 0; i < nsteps; ++i) {
        const int cur = i & 1;
        __syncthreads();
        if (i + 1 < nsteps) stageB((i + 1) * 32, cur ^ 1);
        bfrag_t a  = load_frag(abase + i * 32, DL, lane);
        bfrag_t b0 = load_frag(Bs[cur] + (wn * 32 + 0)  * 32, 32, lane);
        bfrag_t b1 = load_frag(Bs[cur] + (wn * 32 + 16) * 32, 32, lane);
        c0.v = wmma_bf16(a, b0, c0.v);
        c1.v = wmma_bf16(a, b1, c1.v);
    }

    const int hi = lane >> 4, ln = lane & 15;
    for (int sub = 0; sub < 2; ++sub) {
        AccF& cc = sub ? c1 : c0;
        const int n  = tileN * 64 + wn * 32 + sub * 16 + ln;
        const float bv = bo[n];
        for (int r = 0; r < 8; ++r) {
            const int rg = tileM * 64 + wm * 16 + r + 8 * hi;
            const int bb = rg / LPAD;
            const int tp = rg % LPAD;
            if (tp >= PAD)
                out[((long long)bb * L_SEQ + (tp - PAD)) * DM + n] = cc.f[r] + bv;
        }
    }
}

extern "C" void kernel_launch(void* const* d_in, const int* in_sizes, int n_in,
                              void* d_out, int out_size, void* d_ws, size_t ws_size,
                              hipStream_t stream)
{
    const float* q  = (const float*)d_in[0];
    const float* k  = (const float*)d_in[1];
    const float* v  = (const float*)d_in[2];
    const float* Wq = (const float*)d_in[3];
    const float* bq = (const float*)d_in[4];
    const float* Wk = (const float*)d_in[5];
    const float* bk = (const float*)d_in[6];
    const float* Wv = (const float*)d_in[7];
    const float* bv = (const float*)d_in[8];
    const float* Wo = (const float*)d_in[9];
    const float* bo = (const float*)d_in[10];
    float* out = (float*)d_out;

    const int Bn = in_sizes[0] / (L_SEQ * DM);       // 8
    const size_t per = (size_t)Bn * LPAD * DL;       // elements per bf16 buffer
    unsigned short* pq  = (unsigned short*)d_ws;
    unsigned short* pk  = pq  + per;
    unsigned short* pvT = pk  + per;
    unsigned short* po  = pvT + per;

    dim3 blk(256);
    dim3 gProj(Bn * LPAD / 64, DL / 64);
    proj_kernel<<<gProj, blk, 0, stream>>>(q, Wq, bq, pq, 0);
    proj_kernel<<<gProj, blk, 0, stream>>>(k, Wk, bk, pk, 0);
    proj_kernel<<<gProj, blk, 0, stream>>>(v, Wv, bv, pvT, 1);
    attn_kernel<<<dim3(8, NW, Bn), blk, 0, stream>>>(pq, pk, pvT, po);
    outproj_kernel<<<dim3(Bn * LPAD / 64, DM / 64), blk, 0, stream>>>(po, Wo, bo, out);
}
